// W4A16Linear_1176821039707
// MI455X (gfx1250) — matrix operand order repaired and checked by reading, not compiled, not run on hardware
//
#include <hip/hip_runtime.h>
#include <stdint.h>

typedef _Float16 f16;
typedef __attribute__((ext_vector_type(2)))  _Float16 v2h;
typedef __attribute__((ext_vector_type(8)))  _Float16 v8h;
typedef __attribute__((ext_vector_type(16))) _Float16 v16h;
typedef __attribute__((ext_vector_type(8)))  float    v8f;

#define USE_ASYNC_LDS 1

// Problem dimensions (fixed by reference setup_inputs)
constexpr int M_DIM = 4 * 2048;     // B*S flattened rows
constexpr int K_DIM = 4096;         // in features
constexpr int N_DIM = 11008;        // out features
constexpr int KPACK = K_DIM / 2;    // packed int32 (one byte / two nibbles) per out row

constexpr int TILE_M = 256;
constexpr int TILE_N = 128;
constexpr int TILE_K = 64;          // two WMMA K-steps per stage
constexpr int KTILES = K_DIM / TILE_K;  // 64
constexpr int LDSTR  = 80;          // halves per LDS row: 64 data + 16 pad (160B, 32B aligned)

__device__ __forceinline__ void async_cp_b128(f16* lds_dst, const f16* gsrc) {
#if USE_ASYNC_LDS
  // LDS byte offset = low 32 bits of the generic shared-aperture address
  uint32_t lds_off = (uint32_t)(uintptr_t)lds_dst;
  asm volatile("global_load_async_to_lds_b128 %0, %1, off"
               :: "v"(lds_off), "v"(gsrc)
               : "memory");
#else
  *(v8h*)lds_dst = *(const v8h*)gsrc;
#endif
}

__device__ __forceinline__ void wait_async_lds() {
#if USE_ASYNC_LDS
  asm volatile("s_wait_asynccnt 0" ::: "memory");
#endif
}

// Packed int4->fp16 dequant: byte v holds two nibbles.
// u = nibble ^ 8; magic fp16 (0x6400|u) == 1024 + u; result = (u-8)*s via one v_pk_fma_f16.
__device__ __forceinline__ v2h dq2(uint32_t v, v2h s2, v2h b2) {
  uint32_t u = v ^ 0x88u;                                          // flip nibble sign bits
  uint32_t t = (u & 0xFu) | ((u << 12) & 0x000F0000u) | 0x64006400u;
  return __builtin_elementwise_fma(__builtin_bit_cast(v2h, t), s2, b2);
}

__global__ __launch_bounds__(256)
void w4a16_gemm_wmma(const f16* __restrict__ x,
                     const int* __restrict__ wq,
                     const f16* __restrict__ wscale,
                     f16* __restrict__ out)
{
  __shared__ f16 As[2][TILE_M * LDSTR];   // activations, fp16, [m][k]  (80 KB)
  __shared__ f16 Bs[2][TILE_N * LDSTR];   // dequantized weights, fp16, [n][k] (40 KB)

  const int tid  = threadIdx.x;
  const int lane = tid & 31;
  const int wave = tid >> 5;            // 0..7
  const int wm   = wave & 3;            // 4 waves along M (64 rows each)
  const int wn   = wave >> 2;           // 2 waves along N (64 cols each)
  const int laneHalf = lane >> 4;       // 0 | 1
  const int laneIdx  = lane & 15;

  const int mBase = blockIdx.x * TILE_M;
  const int nBase = blockIdx.y * TILE_N;

  // ---- copy mapping ----
  // A tile: 256 rows x 8 b128-chunks = 2048 chunks -> 8 per thread
  // W tile: 128 rows x 8 int4-chunks = 1024 chunks -> 4 per thread
  const int crow = tid >> 3;            // 0..31 (chunk i handles row crow + 32*i)
  const int cidx = tid & 7;             // which 8-half / 4-int chunk in the row

  // Per-channel dequant coefficients (rows fixed over K) -> packed fp16 pairs
  v2h s2[4], b2[4];
#pragma unroll
  for (int i = 0; i < 4; ++i) {
    f16 s  = wscale[nBase + crow + 32 * i];
    f16 bb = s * (f16)(-1032.0f);       // -(1024+8)*s ; 1032 exact in fp16
    s2[i] = (v2h){s, s};
    b2[i] = (v2h){bb, bb};
  }

  auto issue_A = [&](int kb, int buf) {
#pragma unroll
    for (int i = 0; i < 8; ++i) {
      const int row = crow + 32 * i;
      const f16* g = x + (size_t)(mBase + row) * K_DIM + kb * TILE_K + cidx * 8;
      async_cp_b128(&As[buf][row * LDSTR + cidx * 8], g);
    }
  };

  auto load_W = [&](int kb, int4* p) {
#pragma unroll
    for (int i = 0; i < 4; ++i) {
      const int row = crow + 32 * i;
      p[i] = *(const int4*)(wq + (size_t)(nBase + row) * KPACK + kb * (TILE_K / 2) + cidx * 4);
    }
  };

  auto dequant_store = [&](int buf, const int4* p) {
#pragma unroll
    for (int i = 0; i < 4; ++i) {
      const int row = crow + 32 * i;
      const int q[4] = {p[i].x, p[i].y, p[i].z, p[i].w};
      v8h d;
#pragma unroll
      for (int e = 0; e < 4; ++e) {
        v2h r = dq2((uint32_t)q[e], s2[i], b2[i]);
        d[2 * e]     = r[0];
        d[2 * e + 1] = r[1];
      }
      *(v8h*)&Bs[buf][row * LDSTR + cidx * 8] = d;
    }
  };

  v8f acc[4][4] = {};

  auto compute = [&](int buf) {
#pragma unroll
    for (int ks = 0; ks < 2; ++ks) {            // two K=32 WMMA steps per stage
      const int ko = ks * 32;
      v16h a[4], b[4];
#pragma unroll
      for (int i = 0; i < 4; ++i) {
        const f16* row = &As[buf][(wm * 64 + i * 16 + laneIdx) * LDSTR + ko];
        v8h lo = *(const v8h*)(row + laneHalf * 8);
        v8h hi = *(const v8h*)(row + 16 + laneHalf * 8);
        a[i] = __builtin_shufflevector(lo, hi, 0,1,2,3,4,5,6,7,8,9,10,11,12,13,14,15);
      }
#pragma unroll
      for (int j = 0; j < 4; ++j) {
        const f16* row = &Bs[buf][(wn * 64 + j * 16 + laneIdx) * LDSTR + ko];
        v8h lo = *(const v8h*)(row + laneHalf * 8);
        v8h hi = *(const v8h*)(row + 16 + laneHalf * 8);
        b[j] = __builtin_shufflevector(lo, hi, 0,1,2,3,4,5,6,7,8,9,10,11,12,13,14,15);
      }
#pragma unroll
      for (int i = 0; i < 4; ++i)
#pragma unroll
        for (int j = 0; j < 4; ++j)
          acc[i][j] = __builtin_amdgcn_wmma_f32_16x16x32_f16(
              false, a[i], false, b[j], (short)0, acc[i][j], false, false);
    }
  };

  // ---- prologue: fill stage 0 ----
  issue_A(0, 0);
  {
    int4 p[4];
    load_W(0, p);
    dequant_store(0, p);
  }
  wait_async_lds();
  __syncthreads();

  // ---- main K loop, double-buffered ----
  for (int kb = 0; kb < KTILES; ++kb) {
    const int cur = kb & 1;
    const int nxt = cur ^ 1;
    const bool hasNext = (kb + 1 < KTILES);

    int4 p[4];
    if (hasNext) {
      issue_A(kb + 1, nxt);        // async copies overlap WMMA below
      load_W(kb + 1, p);           // global loads stay outstanding through compute
    }

    compute(cur);

    if (hasNext) {
      dequant_store(nxt, p);
      wait_async_lds();
    }
    __syncthreads();
  }

  // ---- epilogue: f32 acc -> f16 out ----
#pragma unroll
  for (int i = 0; i < 4; ++i) {
#pragma unroll
    for (int j = 0; j < 4; ++j) {
      const int m0 = mBase + wm * 64 + i * 16 + laneHalf * 8;
      const int n0 = nBase + wn * 64 + j * 16 + laneIdx;
#pragma unroll
      for (int r = 0; r < 8; ++r)
        out[(size_t)(m0 + r) * N_DIM + n0] = (f16)acc[i][j][r];
    }
  }
}

extern "C" void kernel_launch(void* const* d_in, const int* in_sizes, int n_in,
                              void* d_out, int out_size, void* d_ws, size_t ws_size,
                              hipStream_t stream) {
  (void)in_sizes; (void)n_in; (void)out_size; (void)d_ws; (void)ws_size;
  const f16* x      = (const f16*)d_in[0];
  const int* wq     = (const int*)d_in[1];
  const f16* wscale = (const f16*)d_in[2];
  f16* out          = (f16*)d_out;

  dim3 grid(M_DIM / TILE_M, N_DIM / TILE_N);   // 32 x 86
  dim3 block(256);                             // 8 wave32
  hipLaunchKernelGGL(w4a16_gemm_wmma, grid, block, 0, stream, x, wq, wscale, out);
}
